// AttentionalPropagation_55396488184396
// MI455X (gfx1250) — matrix operand order repaired and checked by reading, not compiled, not run on hardware
//
#include <hip/hip_runtime.h>
#include <cstdint>

// ---------------------------------------------------------------------------
// AttentionalPropagation for MI455X (gfx1250): bf16 WMMA everywhere,
// flash attention (no materialized NxM score matrix), deterministic BN stats,
// TDM (tensor_load_to_lds) staging of the attention V tile.
// ---------------------------------------------------------------------------

#define BATCH   4
#define DMODEL  256
#define NSEQ    4096
#define HEADS   4
#define HDIM    64

typedef __bf16 bf16;
typedef uint32_t u32;
typedef __attribute__((ext_vector_type(16))) __bf16 v16bf;
typedef __attribute__((ext_vector_type(8)))  __bf16 v8bf;
typedef __attribute__((ext_vector_type(8)))  float  v8f;
typedef __attribute__((ext_vector_type(4)))  u32    u32x4;
typedef __attribute__((ext_vector_type(8)))  int    i32x8;
typedef __attribute__((ext_vector_type(4)))  int    i32x4;

#define LDX 40   // LDS leading dim (halves) for [64][32] staging tiles (80B rows, 16B aligned)
#define LDP 72   // LDS leading dim (halves) for [64][64] attention tiles (144B rows, 16B aligned)

#if defined(__has_builtin)
#if __has_builtin(__builtin_amdgcn_tensor_load_to_lds) && __has_builtin(__builtin_amdgcn_s_wait_tensorcnt)
#define HAVE_TDM 1
#endif
#endif
#ifndef HAVE_TDM
#define HAVE_TDM 0
#endif

// --- WMMA fragment loaders (layouts per CDNA5 ISA 7.12.2) -------
// A (16x32 MxK, bf16): lane r=lane%16 holds row r; K chunks [hg*8,+8) and [16+hg*8,+8)
static __device__ __forceinline__ v16bf frag_a_bf(const bf16* tile, int ld) {
  const int lane = threadIdx.x & 31, r = lane & 15, hg = lane >> 4;
  const bf16* row = tile + r * ld + hg * 8;
  v8bf lo = *(const v8bf*)(row);
  v8bf hi = *(const v8bf*)(row + 16);
  return __builtin_shufflevector(lo, hi, 0,1,2,3,4,5,6,7,8,9,10,11,12,13,14,15);
}
// B (32x16 KxN): operand staged as [out-col row][K contiguous]; lane r reads K chunks [hg*8,+8) and [16+hg*8,+8)
static __device__ __forceinline__ v16bf frag_b_bf(const bf16* tile, int ld) {
  const int lane = threadIdx.x & 31, r = lane & 15, hg = lane >> 4;
  const bf16* p = tile + r * ld + hg * 8;
  v8bf lo = *(const v8bf*)(p);
  v8bf hi = *(const v8bf*)(p + 16);
  return __builtin_shufflevector(lo, hi, 0,1,2,3,4,5,6,7,8,9,10,11,12,13,14,15);
}
// A directly from fp32 weights in global (weights are tiny -> L2 resident)
static __device__ __forceinline__ v16bf frag_a_f32(const float* tile, int ld) {
  const int lane = threadIdx.x & 31, r = lane & 15, hg = lane >> 4;
  const float* row = tile + (size_t)r * ld + hg * 8;
  v16bf a;
#pragma unroll
  for (int i = 0; i < 8; ++i) a[i] = (bf16)row[i];
#pragma unroll
  for (int i = 0; i < 8; ++i) a[i + 8] = (bf16)row[16 + i];
  return a;
}
static __device__ __forceinline__ v8f wmma_bf16(v16bf a, v16bf b, v8f c) {
  return __builtin_amdgcn_wmma_f32_16x16x32_bf16(false, a, false, b, (short)0, c, false, false);
}

// ---------------------------------------------------------------------------
// Projection GEMM: Y[b][o][n] (bf16, o-major) = W[OxC] @ X[b][C][N] + bias
// grid (N/64, O/64, B), block 128 (4 waves). X^T tile staged in LDS.
// ---------------------------------------------------------------------------
__global__ void __launch_bounds__(128)
proj_kernel(const float* __restrict__ W, const float* __restrict__ bias,
            const float* __restrict__ X, bf16* __restrict__ Y,
            int C, int O, int Nn) {
  __shared__ __align__(16) bf16 x_s[64 * LDX];
  const int b = blockIdx.z, o0 = blockIdx.y * 64, n0 = blockIdx.x * 64;
  const int tid = threadIdx.x, w = tid >> 5, lane = tid & 31, r = lane & 15, hg = lane >> 4;
  const float* Xb = X + (size_t)b * C * Nn;
  v8f acc[4] = {};
  for (int kk = 0; kk < C; kk += 32) {
    for (int i = tid; i < 64 * 32; i += 128) {
      int n = i & 63, c = i >> 6;
      x_s[n * LDX + c] = (bf16)Xb[(size_t)(kk + c) * Nn + n0 + n];
    }
    __syncthreads();
    v16bf a = frag_a_f32(W + (size_t)(o0 + w * 16) * C + kk, C);
    v16bf bbs[4];
#pragma unroll
    for (int f = 0; f < 4; ++f) bbs[f] = frag_b_bf(x_s + f * 16 * LDX, LDX);
#pragma unroll
    for (int f = 0; f < 4; ++f) acc[f] = wmma_bf16(a, bbs[f], acc[f]);
    __syncthreads();
  }
#pragma unroll
  for (int i = 0; i < 8; ++i) {
    int o = o0 + w * 16 + hg * 8 + i;
    float bv = bias[o];
#pragma unroll
    for (int f = 0; f < 4; ++f) {
      int n = n0 + f * 16 + r;
      Y[((size_t)b * O + o) * Nn + n] = (bf16)(acc[f][i] + bv);
    }
  }
}

// ---------------------------------------------------------------------------
// x -> catT[b][n][c] (first 256 channels of the concat buffer, ld=512)
// ---------------------------------------------------------------------------
__global__ void xpose_kernel(const float* __restrict__ X, bf16* __restrict__ catT) {
  size_t idx = (size_t)blockIdx.x * 256 + threadIdx.x;
  if (idx >= (size_t)BATCH * DMODEL * NSEQ) return;
  int n = (int)(idx & (NSEQ - 1));
  size_t t = idx >> 12;
  int c = (int)(t & (DMODEL - 1));
  int b = (int)(t >> 8);
  catT[((size_t)b * NSEQ + n) * (2 * DMODEL) + c] = (bf16)X[idx];
}

// ---------------------------------------------------------------------------
// Flash attention per (b, h, 64-query tile). Channel map: c = d*HEADS + h.
// V tile staged by the Tensor Data Mover (2D tile, 16B/row LDS padding).
// Writes attnT[b][n][c] bf16 (n-major, ready as B-operand for merge GEMM).
// ---------------------------------------------------------------------------
__global__ void __launch_bounds__(128)
attn_kernel(const bf16* __restrict__ Qp, const bf16* __restrict__ Kp,
            const bf16* __restrict__ Vp, bf16* __restrict__ attnT) {
  __shared__ __align__(16) bf16 q_s[64 * LDP];   // [n][d]
  __shared__ __align__(16) bf16 k_s[64 * LDP];   // [m][d]
  __shared__ __align__(16) bf16 v_s[64 * LDP];   // [d][m]
  __shared__ __align__(16) bf16 p_s[64 * LDP];   // [n][m]
  const int b = blockIdx.z, h = blockIdx.y, n0 = blockIdx.x * 64;
  const int tid = threadIdx.x, w = tid >> 5, lane = tid & 31, r = lane & 15, hg = lane >> 4;
  const size_t plane = (size_t)DMODEL * NSEQ;
  const bf16* Qb = Qp + (size_t)b * plane;
  const bf16* Kb = Kp + (size_t)b * plane;
  const bf16* Vb = Vp + (size_t)b * plane;

  for (int i = tid; i < 64 * 64; i += 128) {
    int n = i & 63, d = i >> 6;
    q_s[n * LDP + d] = Qb[(size_t)(d * HEADS + h) * NSEQ + n0 + n];
  }
  __syncthreads();

  v8f o_acc[4] = {};
  float mrow[8], lrow[8];
#pragma unroll
  for (int i = 0; i < 8; ++i) { mrow[i] = -3.0e38f; lrow[i] = 0.f; }

  const float sc = 0.125f;  // 1/sqrt(64)
  for (int mb = 0; mb < NSEQ / 64; ++mb) {
    const int m0 = mb * 64;
#if HAVE_TDM
    // --- TDM: DMA the 64(d) x 64(m) V tile into v_s with 16B row padding ---
    if (w == 0) {
      uint64_t ga = (uint64_t)(uintptr_t)(Vb + (size_t)h * NSEQ + m0);
      u32 lds_off = (u32)(uintptr_t)v_s;   // low 32 bits of generic addr = LDS offset
      u32x4 g0;
      g0[0] = 1u;                          // count=1, user descriptor
      g0[1] = lds_off;                     // lds_addr
      g0[2] = (u32)ga;                     // global_addr[31:0]
      g0[3] = (u32)((ga >> 32) & 0x01FFFFFFu) | 0x80000000u;  // ga[56:32] | type=2
      i32x8 g1;
      // data_size=2B | pad_enable | pad_interval=4 (pad each 128B) | pad_amount=3 (16B)
      g1[0] = (int)((1u << 16) | (1u << 20) | (4u << 22) | (3u << 25));
      g1[1] = (int)(64u << 16);            // tensor_dim0 = 64 (m)
      g1[2] = (int)(64u << 16);            // tensor_dim1 = 64 (d rows)
      g1[3] = (int)(64u << 16);            // tile_dim0  = 64
      g1[4] = 64;                          // tile_dim1  = 64, tile_dim2 = 0
      g1[5] = HEADS * NSEQ;                // tensor_dim0_stride = 16384 elems
      g1[6] = 0;
      g1[7] = 0;
      i32x4 gz = {0, 0, 0, 0};
#if __clang_major__ >= 23
      i32x8 gz8 = {0, 0, 0, 0, 0, 0, 0, 0};
      __builtin_amdgcn_tensor_load_to_lds(g0, g1, gz, gz, gz8, 0);
#else
      __builtin_amdgcn_tensor_load_to_lds(g0, g1, gz, gz, 0);
#endif
    }
    for (int i = tid; i < 64 * 64; i += 128) {
      int mm = i & 63, d = i >> 6;
      k_s[mm * LDP + d] = Kb[(size_t)(d * HEADS + h) * NSEQ + m0 + mm];
    }
    if (w == 0) __builtin_amdgcn_s_wait_tensorcnt(0);
#else
    for (int i = tid; i < 64 * 64; i += 128) {
      int mm = i & 63, d = i >> 6;
      k_s[mm * LDP + d] = Kb[(size_t)(d * HEADS + h) * NSEQ + m0 + mm];
      v_s[d * LDP + mm] = Vb[(size_t)(d * HEADS + h) * NSEQ + m0 + mm];
    }
#endif
    if (mb + 1 < NSEQ / 64)
      __builtin_prefetch(Kb + (size_t)tid * NSEQ + m0 + 64, 0, 0);
    __syncthreads();

    // S[16n x 64m] per wave = Q^T K
    v8f s[4] = {};
    const bf16* qrow = q_s + (w * 16) * LDP;
#pragma unroll
    for (int kk = 0; kk < 64; kk += 32) {
      v16bf a = frag_a_bf(qrow + kk, LDP);
      v16bf bbs[4];
#pragma unroll
      for (int f = 0; f < 4; ++f) bbs[f] = frag_b_bf(k_s + f * 16 * LDP + kk, LDP);
#pragma unroll
      for (int f = 0; f < 4; ++f) s[f] = wmma_bf16(a, bbs[f], s[f]);
    }

    // online softmax per row (rows hg*8+i, cols across 16 lanes x 4 frags)
#pragma unroll
    for (int i = 0; i < 8; ++i) {
      float mx = -3.0e38f;
#pragma unroll
      for (int f = 0; f < 4; ++f) mx = fmaxf(mx, s[f][i]);
      for (int mask = 1; mask < 16; mask <<= 1) mx = fmaxf(mx, __shfl_xor(mx, mask, 32));
      mx *= sc;
      float newm = fmaxf(mrow[i], mx);
      float corr = __expf(mrow[i] - newm);
      mrow[i] = newm;
      float rsum = 0.f;
#pragma unroll
      for (int f = 0; f < 4; ++f) {
        float p = __expf(s[f][i] * sc - newm);
        rsum += p;
        p_s[(w * 16 + hg * 8 + i) * LDP + f * 16 + r] = (bf16)p;
      }
      for (int mask = 1; mask < 16; mask <<= 1) rsum += __shfl_xor(rsum, mask, 32);
      lrow[i] = lrow[i] * corr + rsum;
#pragma unroll
      for (int f = 0; f < 4; ++f) o_acc[f][i] *= corr;
    }
    asm volatile("s_wait_dscnt 0" ::: "memory");  // wave-private p_s write->read

    // O += P V^T
    const bf16* prow = p_s + (w * 16) * LDP;
#pragma unroll
    for (int kk = 0; kk < 64; kk += 32) {
      v16bf a = frag_a_bf(prow + kk, LDP);
      v16bf bbs[4];
#pragma unroll
      for (int df = 0; df < 4; ++df) bbs[df] = frag_b_bf(v_s + df * 16 * LDP + kk, LDP);
#pragma unroll
      for (int df = 0; df < 4; ++df) o_acc[df] = wmma_bf16(a, bbs[df], o_acc[df]);
    }
    __syncthreads();  // before restaging k_s/v_s
  }

#pragma unroll
  for (int i = 0; i < 8; ++i) {
    float inv = 1.0f / lrow[i];
    int n = n0 + w * 16 + hg * 8 + i;
#pragma unroll
    for (int df = 0; df < 4; ++df) {
      int d = df * 16 + r;
      attnT[((size_t)b * NSEQ + n) * DMODEL + d * HEADS + h] = (bf16)(o_acc[df][i] * inv);
    }
  }
}

// ---------------------------------------------------------------------------
// merge: catT[b][n][256+o] = merge_w @ attnT + merge_b (B-frags straight from global)
// ---------------------------------------------------------------------------
__global__ void __launch_bounds__(128)
merge_kernel(const float* __restrict__ W, const float* __restrict__ bias,
             const bf16* __restrict__ attnT, bf16* __restrict__ catT) {
  const int b = blockIdx.z, o0 = blockIdx.y * 64, n0 = blockIdx.x * 64;
  const int tid = threadIdx.x, w = tid >> 5, lane = tid & 31, r = lane & 15, hg = lane >> 4;
  v8f acc[4] = {};
  for (int kk = 0; kk < DMODEL; kk += 32) {
    v16bf a = frag_a_f32(W + (size_t)(o0 + w * 16) * DMODEL + kk, DMODEL);
    v16bf bbs[4];
#pragma unroll
    for (int f = 0; f < 4; ++f)
      bbs[f] = frag_b_bf(attnT + ((size_t)b * NSEQ + n0 + f * 16) * DMODEL + kk, DMODEL);
#pragma unroll
    for (int f = 0; f < 4; ++f) acc[f] = wmma_bf16(a, bbs[f], acc[f]);
  }
#pragma unroll
  for (int i = 0; i < 8; ++i) {
    int o = o0 + w * 16 + hg * 8 + i;
    float bv = bias[o];
#pragma unroll
    for (int f = 0; f < 4; ++f) {
      int n = n0 + f * 16 + r;
      catT[((size_t)b * NSEQ + n) * (2 * DMODEL) + DMODEL + o] = (bf16)(acc[f][i] + bv);
    }
  }
}

// ---------------------------------------------------------------------------
// mlp1: Hpre[b][c2][n] = mlp1_w @ catT + b1, plus deterministic BN partials
// psum/psumsq layout: [c2][b*64 + ntile]
// ---------------------------------------------------------------------------
__global__ void __launch_bounds__(128)
mlp1_kernel(const float* __restrict__ W, const float* __restrict__ bias,
            const bf16* __restrict__ catT, float* __restrict__ Hpre,
            float* __restrict__ psum, float* __restrict__ psumsq) {
  const int b = blockIdx.z, o0 = blockIdx.y * 64, n0 = blockIdx.x * 64;
  const int tid = threadIdx.x, w = tid >> 5, lane = tid & 31, r = lane & 15, hg = lane >> 4;
  v8f acc[4] = {};
  for (int kk = 0; kk < 2 * DMODEL; kk += 32) {
    v16bf a = frag_a_f32(W + (size_t)(o0 + w * 16) * (2 * DMODEL) + kk, 2 * DMODEL);
    v16bf bbs[4];
#pragma unroll
    for (int f = 0; f < 4; ++f)
      bbs[f] = frag_b_bf(catT + ((size_t)b * NSEQ + n0 + f * 16) * (2 * DMODEL) + kk, 2 * DMODEL);
#pragma unroll
    for (int f = 0; f < 4; ++f) acc[f] = wmma_bf16(a, bbs[f], acc[f]);
  }
  const int nb = b * (NSEQ / 64) + blockIdx.x;
#pragma unroll
  for (int i = 0; i < 8; ++i) {
    int c = o0 + w * 16 + hg * 8 + i;
    float bv = bias[c];
    float s = 0.f, q = 0.f;
#pragma unroll
    for (int f = 0; f < 4; ++f) {
      float v = acc[f][i] + bv;
      Hpre[((size_t)b * (2 * DMODEL) + c) * NSEQ + n0 + f * 16 + r] = v;
      s += v; q += v * v;
    }
    for (int mask = 1; mask < 16; mask <<= 1) {
      s += __shfl_xor(s, mask, 32);
      q += __shfl_xor(q, mask, 32);
    }
    if (r == 0) {
      psum[(size_t)c * (BATCH * NSEQ / 64) + nb] = s;
      psumsq[(size_t)c * (BATCH * NSEQ / 64) + nb] = q;
    }
  }
}

__global__ void bn_stats_kernel(const float* __restrict__ psum, const float* __restrict__ psumsq,
                                const float* __restrict__ gamma, const float* __restrict__ beta,
                                float* __restrict__ bnscale, float* __restrict__ bnshift) {
  int c = blockIdx.x * blockDim.x + threadIdx.x;
  if (c >= 2 * DMODEL) return;
  const int NB = BATCH * NSEQ / 64;
  float s = 0.f, q = 0.f;
  for (int j = 0; j < NB; ++j) { s += psum[(size_t)c * NB + j]; q += psumsq[(size_t)c * NB + j]; }
  const float inv = 1.0f / (float)(BATCH * NSEQ);
  float mean = s * inv;
  float var = q * inv - mean * mean;
  float scl = gamma[c] * rsqrtf(var + 1e-5f);
  bnscale[c] = scl;
  bnshift[c] = beta[c] - mean * scl;
}

// ---------------------------------------------------------------------------
// mlp2: out[b][o][n] f32 = mlp2_w @ relu(bn(Hpre)); BN+ReLU fused into staging
// ---------------------------------------------------------------------------
__global__ void __launch_bounds__(128)
mlp2_kernel(const float* __restrict__ W, const float* __restrict__ bias,
            const float* __restrict__ Hpre, const float* __restrict__ bnscale,
            const float* __restrict__ bnshift, float* __restrict__ out) {
  __shared__ __align__(16) bf16 h_s[64 * LDX];
  const int b = blockIdx.z, o0 = blockIdx.y * 64, n0 = blockIdx.x * 64;
  const int tid = threadIdx.x, w = tid >> 5, lane = tid & 31, r = lane & 15, hg = lane >> 4;
  v8f acc[4] = {};
  for (int kk = 0; kk < 2 * DMODEL; kk += 32) {
    for (int i = tid; i < 64 * 32; i += 128) {
      int n = i & 63, c = i >> 6;
      int ch = kk + c;
      float v = Hpre[((size_t)b * (2 * DMODEL) + ch) * NSEQ + n0 + n];
      v = fmaxf(v * bnscale[ch] + bnshift[ch], 0.f);
      h_s[n * LDX + c] = (bf16)v;
    }
    __syncthreads();
    v16bf a = frag_a_f32(W + (size_t)(o0 + w * 16) * (2 * DMODEL) + kk, 2 * DMODEL);
    v16bf bbs[4];
#pragma unroll
    for (int f = 0; f < 4; ++f) bbs[f] = frag_b_bf(h_s + f * 16 * LDX, LDX);
#pragma unroll
    for (int f = 0; f < 4; ++f) acc[f] = wmma_bf16(a, bbs[f], acc[f]);
    __syncthreads();
  }
#pragma unroll
  for (int i = 0; i < 8; ++i) {
    int o = o0 + w * 16 + hg * 8 + i;
    float bv = bias[o];
#pragma unroll
    for (int f = 0; f < 4; ++f) {
      int n = n0 + f * 16 + r;
      out[((size_t)b * DMODEL + o) * NSEQ + n] = acc[f][i] + bv;
    }
  }
}

// ---------------------------------------------------------------------------
extern "C" void kernel_launch(void* const* d_in, const int* in_sizes, int n_in,
                              void* d_out, int out_size, void* d_ws, size_t ws_size,
                              hipStream_t stream) {
  (void)in_sizes; (void)n_in; (void)out_size; (void)ws_size;
  const float* x       = (const float*)d_in[0];
  const float* source  = (const float*)d_in[1];
  const float* pq_w    = (const float*)d_in[2];
  const float* pq_b    = (const float*)d_in[3];
  const float* pk_w    = (const float*)d_in[4];
  const float* pk_b    = (const float*)d_in[5];
  const float* pv_w    = (const float*)d_in[6];
  const float* pv_b    = (const float*)d_in[7];
  const float* merge_w = (const float*)d_in[8];
  const float* merge_b = (const float*)d_in[9];
  const float* mlp1_w  = (const float*)d_in[10];
  const float* mlp1_b  = (const float*)d_in[11];
  const float* gamma   = (const float*)d_in[12];
  const float* beta    = (const float*)d_in[13];
  const float* mlp2_w  = (const float*)d_in[14];
  const float* mlp2_b  = (const float*)d_in[15];
  float* out = (float*)d_out;

  char* ws = (char*)d_ws;
  size_t off = 0;
  auto alloc = [&](size_t bytes) -> void* {
    void* p = ws + off;
    off += (bytes + 255) & ~(size_t)255;
    return p;
  };
  const size_t planeB = (size_t)BATCH * DMODEL * NSEQ;       // 4M elems
  bf16*  Qp     = (bf16*)alloc(planeB * 2);
  bf16*  Kp     = (bf16*)alloc(planeB * 2);
  bf16*  Vp     = (bf16*)alloc(planeB * 2);
  bf16*  attnT  = (bf16*)alloc(planeB * 2);
  bf16*  catT   = (bf16*)alloc((size_t)BATCH * NSEQ * 2 * DMODEL * 2);
  float* Hpre   = (float*)alloc((size_t)BATCH * 2 * DMODEL * NSEQ * 4);
  float* psum   = (float*)alloc((size_t)2 * DMODEL * (BATCH * NSEQ / 64) * 4);
  float* psumsq = (float*)alloc((size_t)2 * DMODEL * (BATCH * NSEQ / 64) * 4);
  float* bnscale = (float*)alloc(2 * DMODEL * 4);
  float* bnshift = (float*)alloc(2 * DMODEL * 4);

  const dim3 blk(128);
  // q/k/v projections (bf16, channel-major)
  proj_kernel<<<dim3(NSEQ / 64, DMODEL / 64, BATCH), blk, 0, stream>>>(pq_w, pq_b, x,      Qp, DMODEL, DMODEL, NSEQ);
  proj_kernel<<<dim3(NSEQ / 64, DMODEL / 64, BATCH), blk, 0, stream>>>(pk_w, pk_b, source, Kp, DMODEL, DMODEL, NSEQ);
  proj_kernel<<<dim3(NSEQ / 64, DMODEL / 64, BATCH), blk, 0, stream>>>(pv_w, pv_b, source, Vp, DMODEL, DMODEL, NSEQ);
  // x -> first half of concat buffer
  xpose_kernel<<<(int)(planeB / 256), 256, 0, stream>>>(x, catT);
  // flash attention (TDM-staged V tiles)
  attn_kernel<<<dim3(NSEQ / 64, HEADS, BATCH), blk, 0, stream>>>(Qp, Kp, Vp, attnT);
  // merge -> second half of concat buffer
  merge_kernel<<<dim3(NSEQ / 64, DMODEL / 64, BATCH), blk, 0, stream>>>(merge_w, merge_b, attnT, catT);
  // mlp1 + BN partial stats
  mlp1_kernel<<<dim3(NSEQ / 64, 2 * DMODEL / 64, BATCH), blk, 0, stream>>>(mlp1_w, mlp1_b, catT, Hpre, psum, psumsq);
  bn_stats_kernel<<<2, 256, 0, stream>>>(psum, psumsq, gamma, beta, bnscale, bnshift);
  // mlp2 with fused BN+ReLU staging
  mlp2_kernel<<<dim3(NSEQ / 64, DMODEL / 64, BATCH), blk, 0, stream>>>(mlp2_w, mlp2_b, Hpre, bnscale, bnshift, out);
}
